// GRU_55868934586613
// MI455X (gfx1250) — matrix operand order repaired and checked by reading, not compiled, not run on hardware
//
#include <hip/hip_runtime.h>

typedef _Float16 f16;
typedef _Float16 v8h  __attribute__((ext_vector_type(8)));
typedef _Float16 v16h __attribute__((ext_vector_type(16)));
typedef float    v8f  __attribute__((ext_vector_type(8)));

#define TSEQ  2048
#define BATCH 64
#define IDIM  256
#define HDIM  512
#define NWG   32      // workgroups, each owns 16 hidden units
#define WGTH  128     // 4 waves: wave = M-tile (rows 16m..16m+15)
#define NT    96      // gate-column tiles total (1536/16)
#define KC0   24      // k-chunks layer0 ((256+512)/32)
#define KC1   32      // k-chunks layer1 ((512+512)/32)
#define CX0   8       // x-part chunks layer0
#define CX1   16      // x-part chunks layer1

// workspace layout (bytes)
#define OFF_CNT  ((size_t)0)
#define OFF_HB0  ((size_t)256)                                   // [2][64][512] f16 ping-pong
#define OFF_HB1  (OFF_HB0 + (size_t)2*BATCH*HDIM*2)
#define OFF_H1F  (OFF_HB1 + (size_t)2*BATCH*HDIM*2)              // [64][512] f32 final h1
#define OFF_WB0  (OFF_H1F + (size_t)BATCH*HDIM*4)                // packed B frags layer0
#define OFF_WB1  (OFF_WB0 + (size_t)NT*KC0*512*2)
#define OFF_X16  (OFF_WB1 + (size_t)NT*KC1*512*2)                // [T][64][256] f16

// ---------------- helpers ----------------

__device__ __forceinline__ float fast_sigmoid(float x) {
  float e = __builtin_amdgcn_exp2f(-1.4426950408889634f * x);
  return __builtin_amdgcn_rcpf(1.0f + e);
}
__device__ __forceinline__ float fast_tanh(float x) {
  float e = __builtin_amdgcn_exp2f(2.885390081777927f * x); // e^{2x}
  return 1.0f - 2.0f * __builtin_amdgcn_rcpf(e + 1.0f);
}

__device__ __forceinline__ v16h cat16(v8h lo, v8h hb) {
  return __builtin_shufflevector(lo, hb, 0,1,2,3,4,5,6,7,8,9,10,11,12,13,14,15);
}

// A-matrix fragment: lane holds A[m][koff+8*hi .. +7] and A[m][koff+16+8*hi .. +7]
__device__ __forceinline__ v16h load_a(const f16* rowp, int koff, int hi) {
  const f16* p = rowp + koff + 8*hi;
  return cat16(*(const v8h*)p, *(const v8h*)(p + 16));
}
// B fragment pre-packed contiguous (16 halves per lane)
__device__ __forceinline__ v16h load_b(const f16* s) {
  return cat16(*(const v8h*)s, *(const v8h*)(s + 8));
}

__device__ __forceinline__ void grid_sync(unsigned* cnt, unsigned target, int tid) {
  __threadfence();
  __syncthreads();
  if (tid == 0) {
    __hip_atomic_fetch_add(cnt, 1u, __ATOMIC_RELEASE, __HIP_MEMORY_SCOPE_AGENT);
    while (__hip_atomic_load(cnt, __ATOMIC_ACQUIRE, __HIP_MEMORY_SCOPE_AGENT) < target)
      __builtin_amdgcn_s_sleep(1);
  }
  __syncthreads();
  __threadfence();
}

template<int KC, int CX>
__device__ __forceinline__ void phase(const f16* __restrict__ xrow, const f16* __restrict__ hrow,
                                      const f16* __restrict__ sw, int lane,
                                      v8f& ar, v8f& az, v8f& axn, v8f& ahn) {
  ar = {}; az = {}; axn = {}; ahn = {};
  int hi = lane >> 4;
#pragma unroll
  for (int kc = 0; kc < KC; ++kc) {
    const bool isx = (kc < CX);
    v16h a = isx ? load_a(xrow, kc*32, hi) : load_a(hrow, (kc-CX)*32, hi);
    v16h br = load_b(sw + ((size_t)(0*KC + kc)*32 + lane)*16);
    v16h bz = load_b(sw + ((size_t)(1*KC + kc)*32 + lane)*16);
    v16h bn = load_b(sw + ((size_t)(2*KC + kc)*32 + lane)*16);
    ar = __builtin_amdgcn_wmma_f32_16x16x32_f16(false, a, false, br, (short)0, ar, false, false);
    az = __builtin_amdgcn_wmma_f32_16x16x32_f16(false, a, false, bz, (short)0, az, false, false);
    if (isx)
      axn = __builtin_amdgcn_wmma_f32_16x16x32_f16(false, a, false, bn, (short)0, axn, false, false);
    else
      ahn = __builtin_amdgcn_wmma_f32_16x16x32_f16(false, a, false, bn, (short)0, ahn, false, false);
  }
}

__device__ __forceinline__ void gates(const v8f& ar, const v8f& az, const v8f& axn, const v8f& ahn,
                                      float br, float bz, float bxn, float bhn,
                                      float (&hold)[8], f16* __restrict__ hdst,
                                      int rowbase, int colg, float* __restrict__ h1f) {
#pragma unroll
  for (int v = 0; v < 8; ++v) {
    float r = fast_sigmoid(ar[v] + br);
    float z = fast_sigmoid(az[v] + bz);
    float n = fast_tanh(axn[v] + bxn + r * (ahn[v] + bhn));
    float hn = (1.0f - z) * n + z * hold[v];
    hold[v] = hn;
    int rg = rowbase + v;
    hdst[(size_t)rg * HDIM + colg] = (f16)hn;
    if (h1f) h1f[(size_t)rg * HDIM + colg] = hn;
  }
}

// ---------------- kernels ----------------

__global__ void k_init(unsigned* w) {
  size_t i = (size_t)blockIdx.x * 256 + threadIdx.x;
  if (i < OFF_H1F / 4) w[i] = 0u;   // counter + h ping-pong buffers
}

__global__ void k_pack_x(const float* __restrict__ x, f16* __restrict__ x16) {
  size_t g = (size_t)blockIdx.x * 256 + threadIdx.x;
  if (g >= (size_t)TSEQ * BATCH * IDIM) return;
  int i = (int)(g & (IDIM - 1));
  int b = (int)((g >> 8) & 63);
  int t = (int)(g >> 14);
  x16[g] = (f16)x[((size_t)b * TSEQ + t) * IDIM + i];
}

__global__ void k_pack_w(const float* __restrict__ w_ih, const float* __restrict__ w_hh,
                         f16* __restrict__ dst, int KIH, int KC) {
  size_t g = (size_t)blockIdx.x * 256 + threadIdx.x;
  if (g >= (size_t)NT * KC * 512) return;
  int j    = (int)(g & 15);
  int lane = (int)((g >> 4) & 31);
  size_t rem = g >> 9;
  int kc = (int)(rem % KC);
  int nt = (int)(rem / KC);
  int n  = nt * 16 + (lane & 15);
  int k  = kc * 32 + ((j < 8) ? 8 * (lane >> 4) + j : 16 + 8 * (lane >> 4) + (j - 8));       // B frag: K 8h..8h+7 then 16+8h..16+8h+7 per lane-half
  float v = (k < KIH) ? w_ih[(size_t)n * KIH + k]
                      : w_hh[(size_t)n * HDIM + (k - KIH)];
  dst[g] = (f16)v;
}

__global__ __launch_bounds__(WGTH)
void k_gru(const f16* __restrict__ x16,
           const f16* __restrict__ wB0, const f16* __restrict__ wB1,
           const float* __restrict__ b_ih0, const float* __restrict__ b_hh0,
           const float* __restrict__ b_ih1, const float* __restrict__ b_hh1,
           f16* __restrict__ hb0, f16* __restrict__ hb1,
           float* __restrict__ h1fin, unsigned* cnt) {
  __shared__ __align__(16) f16 sW[3*KC0*512 + 3*KC1*512];   // 168 KB of 320 KB WGP LDS

  const int tid = threadIdx.x, wg = blockIdx.x;
  const int lane = tid & 31, mtile = tid >> 5;
  const int lm = lane & 15, hi = lane >> 4;

  // Stage this WG's weight slices into LDS once (gate tile for gate g is global tile g*32+wg)
#pragma unroll
  for (int g = 0; g < 3; ++g) {
    const uint4* src = (const uint4*)(wB0 + (size_t)(g*32 + wg) * KC0 * 512);
    uint4* dst = (uint4*)(sW + (size_t)g * KC0 * 512);
    for (int i = tid; i < KC0*512/8; i += WGTH) dst[i] = src[i];
  }
#pragma unroll
  for (int g = 0; g < 3; ++g) {
    const uint4* src = (const uint4*)(wB1 + (size_t)(g*32 + wg) * KC1 * 512);
    uint4* dst = (uint4*)(sW + (size_t)3*KC0*512 + (size_t)g * KC1 * 512);
    for (int i = tid; i < KC1*512/8; i += WGTH) dst[i] = src[i];
  }
  __syncthreads();

  const int colg = wg * 16 + lm;                 // hidden unit this lane owns
  const int rowbase = mtile * 16 + 8 * hi;       // C-fragment row base
  const float b0r  = b_ih0[colg] + b_hh0[colg];
  const float b0z  = b_ih0[512 + colg] + b_hh0[512 + colg];
  const float b0xn = b_ih0[1024 + colg];
  const float b0hn = b_hh0[1024 + colg];
  const float b1r  = b_ih1[colg] + b_hh1[colg];
  const float b1z  = b_ih1[512 + colg] + b_hh1[512 + colg];
  const float b1xn = b_ih1[1024 + colg];
  const float b1hn = b_hh1[1024 + colg];

  float h0old[8] = {0,0,0,0,0,0,0,0};
  float h1old[8] = {0,0,0,0,0,0,0,0};
  unsigned target = 0;
  const int arow = mtile * 16 + lm;              // A-matrix row this lane loads

  for (int t = 0; t < TSEQ; ++t) {
    // ---- layer 0: A = [x_t ; h0(t-1)] ----
    const f16* xrow0 = x16 + (size_t)t * BATCH * IDIM + (size_t)arow * IDIM;
    const f16* hrow0 = hb0 + (size_t)((t + 1) & 1) * BATCH * HDIM + (size_t)arow * HDIM;
    f16* hW0 = hb0 + (size_t)(t & 1) * BATCH * HDIM;
    v8f ar, az, axn, ahn;
    phase<KC0, CX0>(xrow0, hrow0, sW, lane, ar, az, axn, ahn);
    gates(ar, az, axn, ahn, b0r, b0z, b0xn, b0hn, h0old, hW0, rowbase, colg, nullptr);
    target += NWG; grid_sync(cnt, target, tid);

    // ---- layer 1: A = [h0(t) ; h1(t-1)] ----
    const f16* xrow1 = hW0 + (size_t)arow * HDIM;
    const f16* hrow1 = hb1 + (size_t)((t + 1) & 1) * BATCH * HDIM + (size_t)arow * HDIM;
    f16* hW1 = hb1 + (size_t)(t & 1) * BATCH * HDIM;
    phase<KC1, CX1>(xrow1, hrow1, sW + (size_t)3*KC0*512, lane, ar, az, axn, ahn);
    gates(ar, az, axn, ahn, b1r, b1z, b1xn, b1hn, h1old, hW1, rowbase, colg,
          (t == TSEQ - 1) ? h1fin : nullptr);
    target += NWG; grid_sync(cnt, target, tid);
  }
}

__global__ void k_fc(const float* __restrict__ h1fin, const float* __restrict__ fc_w,
                     const float* __restrict__ fc_b, float* __restrict__ out) {
  int b = threadIdx.x;
  float s = fc_b[0];
  for (int j = 0; j < HDIM; ++j) s += h1fin[(size_t)b * HDIM + j] * fc_w[j];
  out[b] = s;
}

// ---------------- launcher ----------------

extern "C" void kernel_launch(void* const* d_in, const int* in_sizes, int n_in,
                              void* d_out, int out_size, void* d_ws, size_t ws_size,
                              hipStream_t stream) {
  const float* x     = (const float*)d_in[0];
  const float* w_ih0 = (const float*)d_in[1];
  const float* w_hh0 = (const float*)d_in[2];
  const float* b_ih0 = (const float*)d_in[3];
  const float* b_hh0 = (const float*)d_in[4];
  const float* w_ih1 = (const float*)d_in[5];
  const float* w_hh1 = (const float*)d_in[6];
  const float* b_ih1 = (const float*)d_in[7];
  const float* b_hh1 = (const float*)d_in[8];
  const float* fc_w  = (const float*)d_in[9];
  const float* fc_b  = (const float*)d_in[10];

  char* ws = (char*)d_ws;
  unsigned* cnt = (unsigned*)(ws + OFF_CNT);
  f16* hb0   = (f16*)(ws + OFF_HB0);
  f16* hb1   = (f16*)(ws + OFF_HB1);
  float* h1f = (float*)(ws + OFF_H1F);
  f16* wB0   = (f16*)(ws + OFF_WB0);
  f16* wB1   = (f16*)(ws + OFF_WB1);
  f16* x16   = (f16*)(ws + OFF_X16);

  size_t nz = OFF_H1F / 4;
  k_init<<<(unsigned)((nz + 255) / 256), 256, 0, stream>>>((unsigned*)ws);

  size_t nx = (size_t)TSEQ * BATCH * IDIM;
  k_pack_x<<<(unsigned)((nx + 255) / 256), 256, 0, stream>>>(x, x16);

  size_t nw0 = (size_t)NT * KC0 * 512;
  k_pack_w<<<(unsigned)((nw0 + 255) / 256), 256, 0, stream>>>(w_ih0, w_hh0, wB0, IDIM, KC0);
  size_t nw1 = (size_t)NT * KC1 * 512;
  k_pack_w<<<(unsigned)((nw1 + 255) / 256), 256, 0, stream>>>(w_ih1, w_hh1, wB1, HDIM, KC1);

  k_gru<<<NWG, WGTH, 0, stream>>>(x16, wB0, wB1, b_ih0, b_hh0, b_ih1, b_hh1,
                                  hb0, hb1, h1f, cnt);

  k_fc<<<1, BATCH, 0, stream>>>(h1f, fc_w, fc_b, (float*)d_out);
}
